// GraphBinaryClassificationOutputHead_4114578669768
// MI455X (gfx1250) — matrix operand order repaired and checked by reading, not compiled, not run on hardware
//
#include <hip/hip_runtime.h>
#include <hip/hip_bf16.h>

typedef __attribute__((ext_vector_type(16))) __bf16 v16bf;
typedef __attribute__((ext_vector_type(8)))  float  v8f;
typedef __attribute__((ext_vector_type(4)))  float  v4f;

union V16 { v16bf v; uint4 q[2]; };

// ---------------------------------------------------------------------------
// Kernel 0: zero pooled output (atomic target) + split W1 into bf16 hi/lo,
// stored TRANSPOSED as [n][k] (k contiguous) so B-fragments are contiguous.
// ---------------------------------------------------------------------------
__global__ void prep_kernel(const float* __restrict__ W1,
                            float* __restrict__ out,
                            __bf16* __restrict__ wt_hi,
                            __bf16* __restrict__ wt_lo,
                            int B) {
    int idx = blockIdx.x * blockDim.x + threadIdx.x;
    if (idx < 128 * 128) {
        int n = idx >> 7;          // output column
        int k = idx & 127;         // reduction index
        float x = W1[k * 128 + n]; // W1 is [k][n] row-major
        __bf16 h = (__bf16)x;
        wt_hi[idx] = h;
        wt_lo[idx] = (__bf16)(x - (float)h);
    }
    if (idx < B) out[idx] = 0.0f;
}

// ---------------------------------------------------------------------------
// Main kernel: fused  silu(E@W1+b1)@W2+b2  ->  atomic segment-sum.
// 8 waves/block, one 16-row node tile per wave, split-bf16 WMMA (3 products).
// ---------------------------------------------------------------------------
__global__ __launch_bounds__(256)
void GraphHead_mlp_pool_kernel(const float* __restrict__ energy,
                               const int*   __restrict__ batch,
                               const float* __restrict__ b1,
                               const float* __restrict__ W2,
                               const float* __restrict__ b2,
                               const __bf16* __restrict__ wt_hi,
                               const __bf16* __restrict__ wt_lo,
                               float* __restrict__ out,
                               int N) {
    extern __shared__ __bf16 lds[];           // [0,16384): W1T hi ; [16384,32768): W1T lo

    // ---- cooperative stage of W1T hi/lo (64 KB) into LDS, fully coalesced ----
    {
        const uint4* sh = (const uint4*)wt_hi;  // 2048 uint4
        const uint4* sl = (const uint4*)wt_lo;  // 2048 uint4
        uint4* dst = (uint4*)lds;
        int t = threadIdx.x;
        #pragma unroll
        for (int i = 0; i < 8; ++i) {
            dst[t + 256 * i]        = sh[t + 256 * i];
            dst[2048 + t + 256 * i] = sl[t + 256 * i];
        }
    }
    __syncthreads();

    const int lane = threadIdx.x & 31;
    const int wave = threadIdx.x >> 5;
    const int m    = lane & 15;   // row within tile (A) / column index (B,C)
    const int g    = lane >> 4;   // lane-half

    const long tile = (long)blockIdx.x * 8 + wave;
    const long base = tile * 16;
    if (base >= N) return;                     // no barriers past this point

    long row = base + m;
    if (row >= N) row = N - 1;                 // clamp (tail safety; 16 | N anyway)
    const float* arow = energy + (size_t)row * 128;

    const __bf16* ldsh = lds;
    const __bf16* ldsl = lds + 16384;

    v8f acc[8] = {};                           // 8 column tiles of 16 -> full D=128

    #pragma unroll
    for (int ks = 0; ks < 4; ++ks) {           // K window of 32 per step
        const int kwin = ks * 32;

        // ---- A fragment: lane (m, g); elems 0..15 -> K = kwin+16g.. (contiguous)
        //      (matches B fragment k order)
        const v4f* ap0 = (const v4f*)(arow + kwin + 16 * g);
        const v4f* ap1 = (const v4f*)(arow + kwin + 16 * g + 8);
        v4f f0 = __builtin_nontemporal_load(ap0);
        v4f f1 = __builtin_nontemporal_load(ap0 + 1);
        v4f f2 = __builtin_nontemporal_load(ap1);
        v4f f3 = __builtin_nontemporal_load(ap1 + 1);

        float fs[16];
        #pragma unroll
        for (int i = 0; i < 4; ++i) {
            fs[i] = f0[i]; fs[4 + i] = f1[i]; fs[8 + i] = f2[i]; fs[12 + i] = f3[i];
        }

        V16 ah, al;
        #pragma unroll
        for (int e = 0; e < 16; ++e) {
            float x  = fs[e];
            __bf16 h = (__bf16)x;
            ah.v[e] = h;
            al.v[e] = (__bf16)(x - (float)h);
        }

        // ---- per column tile: B fragment from LDS + 3 split-bf16 WMMAs ----
        #pragma unroll
        for (int ct = 0; ct < 8; ++ct) {
            const int nrow = ct * 16 + m;                     // B column (N)
            const int off  = nrow * 128 + kwin + 16 * g;      // element offset
            V16 bh, bl;
            bh.q[0] = *(const uint4*)(ldsh + off);
            bh.q[1] = *(const uint4*)(ldsh + off + 8);
            bl.q[0] = *(const uint4*)(ldsl + off);
            bl.q[1] = *(const uint4*)(ldsl + off + 8);

            acc[ct] = __builtin_amdgcn_wmma_f32_16x16x32_bf16(
                false, ah.v, false, bh.v, (short)0, acc[ct], false, false);
            acc[ct] = __builtin_amdgcn_wmma_f32_16x16x32_bf16(
                false, ah.v, false, bl.v, (short)0, acc[ct], false, false);
            acc[ct] = __builtin_amdgcn_wmma_f32_16x16x32_bf16(
                false, al.v, false, bh.v, (short)0, acc[ct], false, false);
        }
    }

    // ---- epilogue: +b1, SiLU, *W2, reduce over n within each lane-half ----
    float s[8];
    #pragma unroll
    for (int v = 0; v < 8; ++v) s[v] = 0.0f;

    #pragma unroll
    for (int ct = 0; ct < 8; ++ct) {
        const int n  = ct * 16 + m;     // this lane's column in C/D layout
        const float bb = b1[n];
        const float w2 = W2[n];
        #pragma unroll
        for (int v = 0; v < 8; ++v) {
            float c   = acc[ct][v] + bb;
            float sig = 1.0f / (1.0f + __expf(-c));
            s[v] += c * sig * w2;       // silu(c) * W2[n]
        }
    }

    // butterfly reduce within each 16-lane half (masks <= 8 never cross halves)
    #pragma unroll
    for (int v = 0; v < 8; ++v) {
        s[v] += __shfl_xor(s[v], 1);
        s[v] += __shfl_xor(s[v], 2);
        s[v] += __shfl_xor(s[v], 4);
        s[v] += __shfl_xor(s[v], 8);
    }

    const float b2v = b2[0];
    if (m == 0) {                        // lanes 0 (rows 0..7) and 16 (rows 8..15)
        #pragma unroll
        for (int v = 0; v < 8; ++v) {
            long node = base + v + 8 * g;
            if (node < N) {
                int bi = batch[node];
                atomicAdd(out + bi, s[v] + b2v);
            }
        }
    }
}

// ---------------------------------------------------------------------------
extern "C" void kernel_launch(void* const* d_in, const int* in_sizes, int n_in,
                              void* d_out, int out_size, void* d_ws, size_t ws_size,
                              hipStream_t stream) {
    const float* energy = (const float*)d_in[0];
    const int*   batch  = (const int*)  d_in[1];
    const float* W1     = (const float*)d_in[2];
    const float* b1     = (const float*)d_in[3];
    const float* W2     = (const float*)d_in[4];
    const float* b2     = (const float*)d_in[5];
    // d_in[6] = n_molecules (device scalar) == out_size on host

    const int N = in_sizes[0] / 128;
    float* out = (float*)d_out;

    __bf16* wt_hi = (__bf16*)d_ws;       // 128*128 bf16 = 32 KB
    __bf16* wt_lo = wt_hi + 128 * 128;   // 32 KB  (ws usage: 64 KB total)

    int prepN = (out_size > 128 * 128) ? out_size : 128 * 128;
    prep_kernel<<<(prepN + 255) / 256, 256, 0, stream>>>(W1, out, wt_hi, wt_lo, out_size);

    long ntiles  = ((long)N + 15) / 16;
    int  nblocks = (int)((ntiles + 7) / 8);
    GraphHead_mlp_pool_kernel<<<nblocks, 256, 65536, stream>>>(
        energy, batch, b1, W2, b2, wt_hi, wt_lo, out, N);
}
